// CPC_22514218566439
// MI455X (gfx1250) — matrix operand order repaired and checked by reading, not compiled, not run on hardware
//
#include <hip/hip_runtime.h>
#include <hip/hip_bf16.h>

typedef __attribute__((ext_vector_type(16))) __bf16 v16bf;
typedef __attribute__((ext_vector_type(2)))  __bf16 bf16x2;
typedef __attribute__((ext_vector_type(8)))  float  v8f;

#define B_DIM   256
#define L_DIM   512
#define K_OUT   8
#define T_IN    16
#define STEPS   103            // 128 - 16 - (8 + 0 + 1)
#define DENOMF  210944.0f      // B * K * steps

#if __has_builtin(__builtin_amdgcn_global_load_async_to_lds_b128)
#define HAVE_ASYNC_LDS 1
#endif

// The builtin expects pointers to 16-byte int vectors in AS1 (global) / AS3 (LDS).
typedef int v4i_vs __attribute__((vector_size(16)));
typedef __attribute__((address_space(1))) v4i_vs as1_v4i;
typedef __attribute__((address_space(3))) v4i_vs as3_v4i;

__device__ __forceinline__ as1_v4i* gptr16(const void* p) {
  return (as1_v4i*)(unsigned long long)(size_t)p;
}
__device__ __forceinline__ as3_v4i* lptr16(const void* p) {
  // generic LDS address: addr[31:0] is the LDS offset (ISA aperture rule)
  return (as3_v4i*)(unsigned int)(size_t)p;
}

__device__ __forceinline__ void wait_async0() {
#if __has_builtin(__builtin_amdgcn_s_wait_asynccnt)
  __builtin_amdgcn_s_wait_asynccnt(0);
#else
  asm volatile("s_wait_asynccnt 0x0" ::: "memory");
#endif
}

union Frag {
  v16bf v;
  bf16x2 h2[8];
  uint4 q[2];
};

// ---------------------------------------------------------------------------
// Prep: zero accumulators; WkT[k][l][m] = bf16(Wk[k][m][l])
// ---------------------------------------------------------------------------
__global__ void cpc_prep(const float* __restrict__ Wk,
                         __bf16* __restrict__ WkT,
                         float* __restrict__ diag_sum,
                         unsigned int* __restrict__ correct_cnt) {
  unsigned int t = blockIdx.x * blockDim.x + threadIdx.x;
  if (t == 0) *diag_sum = 0.0f;
  if (t == 1) *correct_cnt = 0u;
  const unsigned int total = K_OUT * L_DIM * L_DIM;   // 2^21
  for (unsigned int i = t; i < total; i += gridDim.x * blockDim.x) {
    unsigned int k = i >> 18;
    unsigned int m = (i >> 9) & 511u;
    unsigned int l = i & 511u;
    WkT[(k << 18) + (l << 9) + m] = (__bf16)Wk[i];
  }
}

// ---------------------------------------------------------------------------
// Main: one workgroup per (s, k); 512 threads = 16 waves (wave32).
// sim[b,c] = sum_l V[b,l] * ctx[c,l],  V = enc @ Wk   (bf16 WMMA, f32 acc)
// ---------------------------------------------------------------------------
__global__ __launch_bounds__(512, 1) void cpc_main(
    const float* __restrict__ encoded_x,
    const float* __restrict__ context,
    const __bf16* __restrict__ WkT,
    float* __restrict__ diag_sum,
    unsigned int* __restrict__ correct_cnt) {

  __shared__ __align__(16) __bf16 WkS[64 * 512];   // 64 KB  WkT[l0..l0+64)[m]
  __shared__ __align__(16) __bf16 CtxS[256 * 64];  // 32 KB  ctx[c][l-chunk] bf16
  __shared__ __align__(16) __bf16 VcS[256 * 64];   // 32 KB  V[b][l-chunk] bf16 (reused for argmax)

  const int s    = blockIdx.x;
  const int k    = blockIdx.y;
  const int tid  = threadIdx.x;
  const int w    = tid >> 5;        // wave id 0..15 -> rows [16w, 16w+16)
  const int lane = tid & 31;
  const int half = lane >> 4;
  const int ln   = lane & 15;
  const int b0   = w << 4;

  const float* enc = encoded_x + (size_t)(T_IN + 1 + k + s) * (B_DIM * L_DIM);
  const float* ctx = context   + (size_t)(T_IN + s) * (B_DIM * L_DIM);
  const __bf16* wkt = WkT + ((size_t)k << 18);

  v8f zero = {};
  v8f acc[16];                       // sim rows [b0,b0+16) x all 256 cols
#pragma unroll
  for (int j = 0; j < 16; ++j) acc[j] = zero;

  const float* encRow = enc + (size_t)(b0 + ln) * L_DIM;   // A-fragment row

  for (int lc = 0; lc < 8; ++lc) {
    const int l0 = lc * 64;
    __syncthreads();   // previous iteration's readers of WkS/CtxS/VcS are done

    // --- cooperative: WkT rows [l0,l0+64) -> WkS (contiguous 64 KB copy) ---
#ifdef HAVE_ASYNC_LDS
    {
      const char* src = (const char*)(wkt + (size_t)l0 * L_DIM);
      char* dst = (char*)WkS;
#pragma unroll
      for (int i = 0; i < 8; ++i) {
        int byteoff = (tid + i * 512) * 16;
        __builtin_amdgcn_global_load_async_to_lds_b128(
            gptr16(src + byteoff), lptr16(dst + byteoff), 0, 0);
      }
    }
#else
    {
      const uint4* src = (const uint4*)(wkt + (size_t)l0 * L_DIM);
      uint4* dst = (uint4*)WkS;
#pragma unroll
      for (int i = 0; i < 8; ++i) dst[tid + i * 512] = src[tid + i * 512];
    }
#endif
    // --- cooperative: ctx[:, l0:l0+64) -> CtxS as bf16 (hw pk-cvt) ---
#pragma unroll
    for (int i = 0; i < 8; ++i) {
      int f   = tid + i * 512;                 // float4 index, 4096 total
      int row = f >> 4;
      int c4  = (f & 15) << 2;
      float4 v = *(const float4*)(ctx + (size_t)row * L_DIM + l0 + c4);
      bf16x2 p0 = { (__bf16)v.x, (__bf16)v.y };
      bf16x2 p1 = { (__bf16)v.z, (__bf16)v.w };
      *(bf16x2*)&CtxS[row * 64 + c4]     = p0;
      *(bf16x2*)&CtxS[row * 64 + c4 + 2] = p1;
    }
#ifdef HAVE_ASYNC_LDS
    wait_async0();
#endif
    __syncthreads();

    // --- Stage A: V[b0..b0+16)[l0..l0+64) = enc(16x512) @ Wk(512x64) ---
    v8f accV[4];
#pragma unroll
    for (int j = 0; j < 4; ++j) accV[j] = zero;

    for (int ms = 0; ms < 16; ++ms) {
      const int m0 = ms * 32;
      Frag a;                                   // enc fp32 -> bf16 A-fragment
      {
        const float* pa = encRow + m0 + (half << 3);
        float4 f0 = *(const float4*)(pa);
        float4 f1 = *(const float4*)(pa + 4);
        float4 f2 = *(const float4*)(pa + 16);
        float4 f3 = *(const float4*)(pa + 20);
        a.h2[0] = bf16x2{ (__bf16)f0.x, (__bf16)f0.y };
        a.h2[1] = bf16x2{ (__bf16)f0.z, (__bf16)f0.w };
        a.h2[2] = bf16x2{ (__bf16)f1.x, (__bf16)f1.y };
        a.h2[3] = bf16x2{ (__bf16)f1.z, (__bf16)f1.w };
        a.h2[4] = bf16x2{ (__bf16)f2.x, (__bf16)f2.y };
        a.h2[5] = bf16x2{ (__bf16)f2.z, (__bf16)f2.w };
        a.h2[6] = bf16x2{ (__bf16)f3.x, (__bf16)f3.y };
        a.h2[7] = bf16x2{ (__bf16)f3.z, (__bf16)f3.w };
      }
#pragma unroll
      for (int j = 0; j < 4; ++j) {
        const int off = (j * 16 + ln) * L_DIM + m0 + (half << 3);
        Frag b;
        b.q[0] = *(const uint4*)&WkS[off];
        b.q[1] = *(const uint4*)&WkS[off + 16];
        accV[j] = __builtin_amdgcn_wmma_f32_16x16x32_bf16(
            false, a.v, false, b.v, (short)0, accV[j], false, false);
      }
    }
    // write V chunk to LDS as bf16
#pragma unroll
    for (int j = 0; j < 4; ++j) {
#pragma unroll
      for (int r = 0; r < 8; ++r) {
        int row = b0 + r + (half << 3);
        int col = j * 16 + ln;
        VcS[row * 64 + col] = (__bf16)accV[j][r];
      }
    }
    __syncthreads();

    // --- Stage B: acc += V(16x64) @ ctx_chunk^T (64-deep, 2 K-steps) ---
#pragma unroll
    for (int ks = 0; ks < 2; ++ks) {
      const int kl = ks * 32;
      Frag a;
      const __bf16* pva = &VcS[(b0 + ln) * 64 + kl + (half << 3)];
      a.q[0] = *(const uint4*)(pva);
      a.q[1] = *(const uint4*)(pva + 16);
#pragma unroll
      for (int j = 0; j < 16; ++j) {
        const int off = (j * 16 + ln) * 64 + kl + (half << 3);
        Frag b;
        b.q[0] = *(const uint4*)&CtxS[off];
        b.q[1] = *(const uint4*)&CtxS[off + 16];
        acc[j] = __builtin_amdgcn_wmma_f32_16x16x32_bf16(
            false, a.v, false, b.v, (short)0, acc[j], false, false);
      }
    }
  }
  __syncthreads();   // done with VcS as V-tile; reuse for argmax scratch

  // --- row-wise logsumexp: wave owns full rows; D element (m=r+8*half, n=ln)
  float lse[8];
#pragma unroll
  for (int r = 0; r < 8; ++r) {
    float mx = acc[0][r];
#pragma unroll
    for (int j = 1; j < 16; ++j) mx = fmaxf(mx, acc[j][r]);
#pragma unroll
    for (int o = 1; o < 16; o <<= 1) mx = fmaxf(mx, __shfl_xor(mx, o, 32));
    float sm = 0.0f;
#pragma unroll
    for (int j = 0; j < 16; ++j) sm += expf(acc[j][r] - mx);
#pragma unroll
    for (int o = 1; o < 16; o <<= 1) sm += __shfl_xor(sm, o, 32);
    lse[r] = mx + logf(sm);
  }

  // --- diagonal logp sum: diagonal of this wave's rows lives in tile j == w
  float dsum = 0.0f;
#pragma unroll
  for (int j = 0; j < 16; ++j) {
    if (j == w) {
#pragma unroll
      for (int r = 0; r < 8; ++r) {
        if (lane == r || lane == r + 24) dsum += acc[j][r] - lse[r];
      }
    }
  }
#pragma unroll
  for (int o = 1; o < 32; o <<= 1) dsum += __shfl_xor(dsum, o, 32);
  if (lane == 0) atomicAdd(diag_sum, dsum);

  // --- column argmax over b of (sim[b,c] - lse[b]); first-occurrence ties ---
  float* argV = (float*)VcS;                    // [16][256]
  int*   argI = (int*)((char*)VcS + 16384);     // [16][256]
#pragma unroll
  for (int j = 0; j < 16; ++j) {
    float bv = -__builtin_inff();
    int   bb = 0x7fffffff;
#pragma unroll
    for (int r = 0; r < 8; ++r) {
      float v = acc[j][r] - lse[r];
      int   b = b0 + r + (half << 3);
      if (v > bv || (v == bv && b < bb)) { bv = v; bb = b; }
    }
    float pv = __shfl_xor(bv, 16, 32);          // combine the two half-wave row sets
    int   pb = __shfl_xor(bb, 16, 32);
    if (pv > bv || (pv == bv && pb < bb)) { bv = pv; bb = pb; }
    if (lane < 16) {
      argV[w * 256 + j * 16 + ln] = bv;
      argI[w * 256 + j * 16 + ln] = bb;
    }
  }
  __syncthreads();

  if (tid < 256) {                              // waves 0..7, fully active
    const int c = tid;
    float bv = argV[c];
    int   bb = argI[c];
#pragma unroll
    for (int ww = 1; ww < 16; ++ww) {
      float v = argV[ww * 256 + c];
      int   b = argI[ww * 256 + c];
      if (v > bv || (v == bv && b < bb)) { bv = v; bb = b; }
    }
    unsigned long long m = __ballot(bb == c);
    if (lane == 0) atomicAdd(correct_cnt, (unsigned int)__popcll(m));
  }
}

// ---------------------------------------------------------------------------
__global__ void cpc_final(const float* __restrict__ diag_sum,
                          const unsigned int* __restrict__ correct_cnt,
                          float* __restrict__ out) {
  if (blockIdx.x == 0 && threadIdx.x == 0) {
    out[0] = (float)(*correct_cnt) / DENOMF;    // accuracy
    out[1] = (*diag_sum) / (-DENOMF);           // loss
  }
}

// ---------------------------------------------------------------------------
extern "C" void kernel_launch(void* const* d_in, const int* in_sizes, int n_in,
                              void* d_out, int out_size, void* d_ws, size_t ws_size,
                              hipStream_t stream) {
  const float* encoded_x = (const float*)d_in[0];
  const float* context   = (const float*)d_in[1];
  const float* Wk        = (const float*)d_in[2];
  float* out = (float*)d_out;

  float*        diag = (float*)d_ws;
  unsigned int* cnt  = (unsigned int*)((char*)d_ws + 64);
  __bf16*       WkT  = (__bf16*)((char*)d_ws + 256);
  if (ws_size < (size_t)256 + (size_t)K_OUT * L_DIM * L_DIM * 2) return; // ~4.2 MB

  cpc_prep<<<2048, 256, 0, stream>>>(Wk, WkT, diag, cnt);
  dim3 grid(STEPS, K_OUT);
  cpc_main<<<grid, 512, 0, stream>>>(encoded_x, context, WkT, diag, cnt);
  cpc_final<<<1, 32, 0, stream>>>(diag, cnt, out);
}
